// MiddleMENoDownsampleXYMaxPool_39685497815878
// MI455X (gfx1250) — compile-verified, matrix operand order checked by reading, not run on hardware
//
#include <hip/hip_runtime.h>

typedef __bf16 bf16;
typedef __attribute__((ext_vector_type(16))) __bf16 v16bf;
typedef __attribute__((ext_vector_type(8)))  float  v8f;

struct __align__(16) U4 { unsigned int x, y, z, w; };

#define HGT 128
#define WID 128
#define NEGF (-1e30f)

// ---------------------------------------------------------------- utilities
__global__ void zero_u4_kernel(U4* __restrict__ p, long n) {
  long i = (long)blockIdx.x * blockDim.x + threadIdx.x;
  if (i < n) { U4 z = {0u, 0u, 0u, 0u}; p[i] = z; }
}

// scatter voxel features (NPTS x 16 f32) into dense bf16 grid [d][h][w][16] + occ mask
__global__ void scatter_kernel(const float* __restrict__ vf, const int* __restrict__ coors,
                               bf16* __restrict__ grid, unsigned char* __restrict__ occ, int npts) {
  int i = blockIdx.x * blockDim.x + threadIdx.x;
  if (i >= npts) return;
  int d = coors[i * 4 + 1], h = coors[i * 4 + 2], w = coors[i * 4 + 3];
  long pos = ((long)d * HGT + h) * WID + w;
  occ[pos] = 1;
#pragma unroll
  for (int c = 0; c < 16; ++c) grid[pos * 16 + c] = (bf16)vf[i * 16 + c];
}

// Repack OIDHW f32 weights into WMMA bf16 A-fragment layout:
// block index = (coutTile*27 + tap)*KCHUNK + kc ; per block: 32 lanes x 16 halves.
// Lane layout (16-bit A 16x32): M = lane&15 ; K(p) = ((p<8)?p:p+8) + (lane<16?0:8).
__global__ void repack_kernel(const float* __restrict__ wsrc, bf16* __restrict__ pw,
                              int Cout, int Cin, int KCHUNK) {
  int gid = blockIdx.x * blockDim.x + threadIdx.x;
  int total = (Cout / 16) * 27 * KCHUNK * 32;
  if (gid >= total) return;
  int lane = gid & 31;
  int blk  = gid >> 5;
  int kc   = blk % KCHUNK;
  int tap  = (blk / KCHUNK) % 27;
  int tile = blk / (KCHUNK * 27);
  int M  = lane & 15;
  int kb = (lane < 16) ? 0 : 8;
  union { U4 q[2]; bf16 hb[16]; } st;
#pragma unroll
  for (int p = 0; p < 16; ++p) {
    int K = ((p < 8) ? p : p + 8) + kb;
    int cin = kc * 32 + K;
    float v = 0.f;
    if (cin < Cin) v = wsrc[((long)(tile * 16 + M) * Cin + cin) * 27 + tap];
    st.hb[p] = (bf16)v;
  }
  U4* dst = (U4*)(pw + (long)blk * 512 + lane * 16);
  dst[0] = st.q[0];
  dst[1] = st.q[1];
}

// ---------------------------------------------------------------- implicit-GEMM conv3d (3x3x3, SAME)
// in  : [DD][128][128][CIN]  bf16 (zeros outside occ)
// out : [DD][128][128][COUT] bf16, masked ReLU by occ
// pw  : packed weights (see repack_kernel)
template <int CIN, int COUT>
__global__ __launch_bounds__((COUT / 16) * 32)
void conv3d_wmma_kernel(const bf16* __restrict__ in, bf16* __restrict__ out,
                        const unsigned char* __restrict__ occ,
                        const bf16* __restrict__ pw, int DD) {
  constexpr int KPAD = ((CIN + 31) / 32) * 32;   // K padded to WMMA depth
  constexpr int KCH  = KPAD / 32;                // 16x16x32 steps per tap
  constexpr int NW   = COUT / 16;                // one wave per 16-Cout tile
  constexpr int CH8  = KPAD / 8;                 // 16B chunks per LDS row
  __shared__ __align__(16) bf16 smem[3 * 3 * 18 * KPAD];

  const int w0  = blockIdx.x * 16;
  const int h0  = blockIdx.y;
  const int d0  = blockIdx.z;
  const int tid = threadIdx.x;

  // stage 3x3x18 halo (zero-padded) into LDS, channel-contiguous rows of KPAD
  for (int i = tid; i < 162 * CH8; i += NW * 32) {
    int c8  = i % CH8;
    int pos = i / CH8;
    int wz = pos % 18, hz = (pos / 18) % 3, dz = pos / 54;
    int d = d0 + dz - 1, h = h0 + hz - 1, w = w0 + wz - 1;
    U4 v = {0u, 0u, 0u, 0u};
    if (d >= 0 && d < DD && h >= 0 && h < HGT && w >= 0 && w < WID && c8 * 8 < CIN)
      v = *(const U4*)(in + (((long)d * HGT + h) * WID + w) * CIN + c8 * 8);
    *(U4*)(smem + pos * KPAD + c8 * 8) = v;
  }
  __syncthreads();

  const int wave = tid >> 5;
  const int lane = tid & 31;
  const int n    = lane & 15;               // output position within 16-wide tile (N)
  const int kb   = (lane < 16) ? 0 : 8;     // B/A fragment K base per lane half

  v8f acc = {0.f, 0.f, 0.f, 0.f, 0.f, 0.f, 0.f, 0.f};
  const bf16* pwt = pw + (long)wave * 27 * KCH * 512;

  for (int tap = 0; tap < 27; ++tap) {
    const int kd = tap / 9, kh = (tap / 3) % 3, kw = tap % 3;
    const bf16* lrow = smem + ((kd * 3 + kh) * 18 + (n + kw)) * KPAD;
    const bf16* arow = pwt + (long)tap * KCH * 512 + lane * 16;
#pragma unroll
    for (int kc = 0; kc < KCH; ++kc) {
      union { v16bf v; U4 q[2]; } A, B;
      const bf16* ap = arow + kc * 512;      // 32B contiguous per lane (packed layout)
      A.q[0] = *(const U4*)(ap);
      A.q[1] = *(const U4*)(ap + 8);
      const bf16* bp = lrow + kc * 32 + kb;  // K in {kb..kb+7} and {kb+16..kb+23}
      B.q[0] = *(const U4*)(bp);
      B.q[1] = *(const U4*)(bp + 16);
      acc = __builtin_amdgcn_wmma_f32_16x16x32_bf16(false, A.v, false, B.v,
                                                    (short)0, acc, false, false);
    }
  }

  // epilogue: C/D layout -> lane = N (spatial), VGPR row r = M (cout offset, +8 for hi lanes)
  const int  w    = w0 + n;
  const long opos = ((long)d0 * HGT + h0) * WID + w;
  const bool ov   = occ[opos] != 0;
  const int  hi   = (lane < 16) ? 0 : 8;
  union { U4 q; bf16 hb[8]; } st;
#pragma unroll
  for (int r = 0; r < 8; ++r) {
    float v = acc[r];
    v = (ov && v > 0.f) ? v : 0.f;           // masked ReLU
    st.hb[r] = (bf16)v;
  }
  *(U4*)(out + opos * COUT + wave * 16 + hi) = st.q;
}

// ---------------------------------------------------------------- max pools (C = 64)
// window 3x3x3, stride 2 in D, pad 1; occ masked with NEG; occ_out = occ[2d]|occ[2d+1]
__global__ void pool_spatial_kernel(const bf16* __restrict__ in, const unsigned char* __restrict__ occ_in,
                                    bf16* __restrict__ out, unsigned char* __restrict__ occ_out, int Din) {
  const int C8 = 8;
  long gid = (long)blockIdx.x * blockDim.x + threadIdx.x;
  long total = (long)(Din / 2) * HGT * WID * C8;
  if (gid >= total) return;
  int  c8  = (int)(gid % C8);
  long pos = gid / C8;
  int w  = (int)(pos % WID);
  int h  = (int)((pos / WID) % HGT);
  int dd = (int)(pos / ((long)WID * HGT));
  float m[8];
#pragma unroll
  for (int r = 0; r < 8; ++r) m[r] = NEGF;
  for (int dz = 2 * dd - 1; dz <= 2 * dd + 1; ++dz) {
    if (dz < 0 || dz >= Din) continue;
    for (int hz = h - 1; hz <= h + 1; ++hz) {
      if (hz < 0 || hz >= HGT) continue;
      for (int wz = w - 1; wz <= w + 1; ++wz) {
        if (wz < 0 || wz >= WID) continue;
        long sp = ((long)dz * HGT + hz) * WID + wz;
        if (!occ_in[sp]) continue;
        U4 q = *(const U4*)(in + sp * 64 + c8 * 8);
        const bf16* hb = (const bf16*)&q;
#pragma unroll
        for (int r = 0; r < 8; ++r) { float v = (float)hb[r]; m[r] = m[r] > v ? m[r] : v; }
      }
    }
  }
  long s0 = ((long)(2 * dd) * HGT + h) * WID + w;
  long s1 = ((long)(2 * dd + 1) * HGT + h) * WID + w;
  bool o2 = occ_in[s0] || occ_in[s1];
  if (c8 == 0) occ_out[pos] = o2 ? 1 : 0;
  union { U4 q; bf16 hb[8]; } st;
#pragma unroll
  for (int r = 0; r < 8; ++r) st.hb[r] = (bf16)(o2 ? m[r] : 0.f);
  *(U4*)(out + pos * 64 + c8 * 8) = st.q;
}

// window 3x1x1, stride 2 in D, pad 1 in D only
__global__ void pool_d_kernel(const bf16* __restrict__ in, const unsigned char* __restrict__ occ_in,
                              bf16* __restrict__ out, unsigned char* __restrict__ occ_out, int Din) {
  const int C8 = 8;
  long gid = (long)blockIdx.x * blockDim.x + threadIdx.x;
  long total = (long)(Din / 2) * HGT * WID * C8;
  if (gid >= total) return;
  int  c8  = (int)(gid % C8);
  long pos = gid / C8;
  int w  = (int)(pos % WID);
  int h  = (int)((pos / WID) % HGT);
  int dd = (int)(pos / ((long)WID * HGT));
  float m[8];
#pragma unroll
  for (int r = 0; r < 8; ++r) m[r] = NEGF;
  for (int dz = 2 * dd - 1; dz <= 2 * dd + 1; ++dz) {
    if (dz < 0 || dz >= Din) continue;
    long sp = ((long)dz * HGT + h) * WID + w;
    if (!occ_in[sp]) continue;
    U4 q = *(const U4*)(in + sp * 64 + c8 * 8);
    const bf16* hb = (const bf16*)&q;
#pragma unroll
    for (int r = 0; r < 8; ++r) { float v = (float)hb[r]; m[r] = m[r] > v ? m[r] : v; }
  }
  long s0 = ((long)(2 * dd) * HGT + h) * WID + w;
  long s1 = ((long)(2 * dd + 1) * HGT + h) * WID + w;
  bool o2 = occ_in[s0] || occ_in[s1];
  if (c8 == 0) occ_out[pos] = o2 ? 1 : 0;
  union { U4 q; bf16 hb[8]; } st;
#pragma unroll
  for (int r = 0; r < 8; ++r) st.hb[r] = (bf16)(o2 ? m[r] : 0.f);
  *(U4*)(out + pos * 64 + c8 * 8) = st.q;
}

// act [2][128][128][64] bf16 -> out (1, 64*2, 128, 128) f32 ; channel = c*2 + dd
__global__ void finalize_kernel(const bf16* __restrict__ act, float* __restrict__ out) {
  long gid = (long)blockIdx.x * blockDim.x + threadIdx.x;
  const long total = 2L * HGT * WID * 64;
  if (gid >= total) return;
  int  c   = (int)(gid % 64);
  long pos = gid / 64;
  int w  = (int)(pos % WID);
  int h  = (int)((pos / WID) % HGT);
  int dd = (int)(pos / ((long)WID * HGT));
  out[((long)(c * 2 + dd) * HGT + h) * WID + w] = (float)act[gid];
}

// ---------------------------------------------------------------- launcher
extern "C" void kernel_launch(void* const* d_in, const int* in_sizes, int n_in,
                              void* d_out, int out_size, void* d_ws, size_t ws_size,
                              hipStream_t stream) {
  (void)n_in; (void)out_size; (void)ws_size;
  const float* vf    = (const float*)d_in[0];
  const int*   coors = (const int*)d_in[1];
  const float* wts[8];
  for (int i = 0; i < 8; ++i) wts[i] = (const float*)d_in[3 + i];
  const int npts = in_sizes[0] / 16;

  char* base = (char*)d_ws;
  size_t off = 0;
  auto carve = [&](size_t bytes) -> char* {
    char* p = base + off;
    off += (bytes + 255) & ~(size_t)255;
    return p;
  };
  bf16* actA = (bf16*)carve(16L * HGT * WID * 64 * 2);  // ping
  bf16* actB = (bf16*)carve(16L * HGT * WID * 64 * 2);  // pong
  unsigned char* occ0 = (unsigned char*)carve(16L * HGT * WID);
  unsigned char* occ1 = (unsigned char*)carve(8L * HGT * WID);
  unsigned char* occ2 = (unsigned char*)carve(4L * HGT * WID);
  unsigned char* occ3 = (unsigned char*)carve(2L * HGT * WID);
  bf16* pw1 = (bf16*)carve(2L * 27 * 1 * 512 * 2);      // 32x16
  bf16* pw2 = (bf16*)carve(4L * 27 * 1 * 512 * 2);      // 64x32
  bf16* pw64[6];
  for (int i = 0; i < 6; ++i) pw64[i] = (bf16*)carve(4L * 27 * 2 * 512 * 2);  // 64x64

  // zero layer-1 input grid (16ch region) and occ mask each call (deterministic)
  { long n = (16L * HGT * WID * 16 * 2) / 16;
    zero_u4_kernel<<<(int)((n + 255) / 256), 256, 0, stream>>>((U4*)actA, n); }
  { long n = (16L * HGT * WID) / 16;
    zero_u4_kernel<<<(int)((n + 255) / 256), 256, 0, stream>>>((U4*)occ0, n); }

  scatter_kernel<<<(npts + 255) / 256, 256, 0, stream>>>(vf, coors, actA, occ0, npts);

  auto repack = [&](const float* w, bf16* p, int Cout, int Cin) {
    int kch = (Cin + 31) / 32;
    int total = (Cout / 16) * 27 * kch * 32;
    repack_kernel<<<(total + 255) / 256, 256, 0, stream>>>(w, p, Cout, Cin, kch);
  };
  repack(wts[0], pw1, 32, 16);
  repack(wts[1], pw2, 64, 32);
  for (int i = 0; i < 6; ++i) repack(wts[2 + i], pw64[i], 64, 64);

  // stage 1: D=16
  conv3d_wmma_kernel<16, 32><<<dim3(8, HGT, 16),  64, 0, stream>>>(actA, actB, occ0, pw1, 16);
  conv3d_wmma_kernel<32, 64><<<dim3(8, HGT, 16), 128, 0, stream>>>(actB, actA, occ0, pw2, 16);
  { long n = 8L * HGT * WID * 8;
    pool_spatial_kernel<<<(int)((n + 255) / 256), 256, 0, stream>>>(actA, occ0, actB, occ1, 16); }

  // stage 2: D=8
  conv3d_wmma_kernel<64, 64><<<dim3(8, HGT, 8), 128, 0, stream>>>(actB, actA, occ1, pw64[0], 8);
  conv3d_wmma_kernel<64, 64><<<dim3(8, HGT, 8), 128, 0, stream>>>(actA, actB, occ1, pw64[1], 8);
  conv3d_wmma_kernel<64, 64><<<dim3(8, HGT, 8), 128, 0, stream>>>(actB, actA, occ1, pw64[2], 8);
  { long n = 4L * HGT * WID * 8;
    pool_spatial_kernel<<<(int)((n + 255) / 256), 256, 0, stream>>>(actA, occ1, actB, occ2, 8); }

  // stage 3: D=4
  conv3d_wmma_kernel<64, 64><<<dim3(8, HGT, 4), 128, 0, stream>>>(actB, actA, occ2, pw64[3], 4);
  conv3d_wmma_kernel<64, 64><<<dim3(8, HGT, 4), 128, 0, stream>>>(actA, actB, occ2, pw64[4], 4);
  conv3d_wmma_kernel<64, 64><<<dim3(8, HGT, 4), 128, 0, stream>>>(actB, actA, occ2, pw64[5], 4);
  { long n = 2L * HGT * WID * 8;
    pool_d_kernel<<<(int)((n + 255) / 256), 256, 0, stream>>>(actA, occ2, actB, occ3, 4); }

  // reshape to (1, 128, 128, 128) f32
  { long n = 2L * HGT * WID * 64;
    finalize_kernel<<<(int)((n + 255) / 256), 256, 0, stream>>>(actB, (float*)d_out); }
}
